// HashGenerator_69818988364216
// MI455X (gfx1250) — matrix operand order repaired and checked by reading, not compiled, not run on hardware
//
#include <hip/hip_runtime.h>
#include <hip/hip_bf16.h>
#include <math.h>

typedef __attribute__((ext_vector_type(16))) _Float16 v16h;
typedef __attribute__((ext_vector_type(8)))  _Float16 v8h;
typedef __attribute__((ext_vector_type(8)))  float    v8f;
typedef __attribute__((ext_vector_type(4)))  unsigned int v4u;
typedef __attribute__((ext_vector_type(8)))  int      v8i;
typedef __attribute__((ext_vector_type(4)))  int      v4i;

#define DTAB   131072      // 16 * 4096 * 2
#define ZDIM   512
#define NLVL   16
#define TSIZE  4096
#define IMGN   65536       // 256*256
#define NBATCH 8
#define KCHUNK 32          // K per pipeline stage (one WMMA k-step)
#define NCHUNK 16          // 512 / 32

// ---- padded LDS strides (halves), multiples of 8 halves (16B), chosen so the
// ---- 16-lane b128 fragment loads land on 64 distinct banks.
#define LZ_STRIDE 520
#define W1S 40
#define W2S 72
#define W3S 72
#define HS  72

__device__ __forceinline__ v16h ld16h(const _Float16* p) {
  union { v16h v; v8h h[2]; } u;
  u.h[0] = *(const v8h*)(p);
  u.h[1] = *(const v8h*)(p + 8);
  return u.v;
}

// Issue one per-wave TDM: rows [kb + wv*4, +4), cols [col0, +128) -> LDS.
__device__ __forceinline__ void tdm_issue(const float* wt, int col0, int kb,
                                          int wv, unsigned lds_addr) {
  const unsigned long long ga =
      (unsigned long long)(size_t)(wt + (size_t)(kb + wv * 4) * DTAB + col0);
  v4u g0;
  g0.x = 1u;                                           // count = 1
  g0.y = lds_addr + (unsigned)(wv * 4 * 128 * 4);      // lds_addr (bytes)
  g0.z = (unsigned)(ga & 0xffffffffu);                 // global_addr lo
  g0.w = (unsigned)((ga >> 32) & 0x01ffffffu)          // global_addr hi
         | (2u << 30);                                 // type = 2 (image)
  v8i g1;
  g1[0] = 0x00020000;                // wg_mask=0, data_size=2 (4 bytes)
  g1[1] = 0;                         // atomic barrier off, td0[15:0]=0
  g1[2] = (int)((512u << 16) | 2u);  // tensor_dim1=512 | tensor_dim0 hi16
  g1[3] = (int)(128u << 16);         // tile_dim0 = 128 (contiguous cols)
  g1[4] = 4;                         // tile_dim1 = 4 rows/wave
  g1[5] = (int)DTAB;                 // tensor_dim0_stride = 131072 (lo32)
  g1[6] = 0;
  g1[7] = 0;
  const v4i g2 = {0, 0, 0, 0};       // 2-D tensor: groups 2/3 unused
  const v4i g3 = {0, 0, 0, 0};
  const v8i g4 = {0, 0, 0, 0, 0, 0, 0, 0};
  __builtin_amdgcn_tensor_load_to_lds(g0, g1, g2, g3, g4, 0);
}

// ============================================================================
// Kernel A: tables = z(8x512) @ w_table(512x131072) + b_table
// Double-buffered TDM pipeline: tensor_load_to_lds streams f32 w_table chunks
// into ping-pong LDS buffers; chunk i+1's DMA overlaps chunk i's WMMA work.
// grid 1024 blocks x 256 thr; each block owns 128 columns, K chunked by 32.
// ============================================================================
__global__ __launch_bounds__(256)
void k_tables(const float* __restrict__ z, const float* __restrict__ wt,
              const float* __restrict__ bt, float* __restrict__ tab) {
  __shared__ _Float16 lz[16 * LZ_STRIDE];        // z padded to 16 rows, f16
  __shared__ float    raw[2][KCHUNK * 128];      // ping-pong TDM targets
  const int tid  = threadIdx.x;
  const int col0 = blockIdx.x * 128;
  const int lane = tid & 31, wv = tid >> 5;
  const int n = lane & 15, hf = lane >> 4;
  const unsigned lds0 = (unsigned)(size_t)(void*)&raw[0][0];
  const unsigned lds1 = (unsigned)(size_t)(void*)&raw[1][0];

  // kick off chunk 0 DMA first, so it overlaps the z staging below
  tdm_issue(wt, col0, 0, wv, lds0);

  // stage z -> LDS [m][k], rows 8..15 zero (batch pad)
  for (int idx = tid; idx < 16 * ZDIM; idx += 256) {
    const int m = idx >> 9, k = idx & (ZDIM - 1);
    lz[m * LZ_STRIDE + k] = (m < NBATCH) ? (_Float16)z[m * ZDIM + k] : (_Float16)0.0f;
  }

  __builtin_amdgcn_s_wait_tensorcnt(0);
  __syncthreads();                               // chunk 0 + lz visible

  v8f acc = {};
  for (int i = 0; i < NCHUNK; ++i) {
    const int kb = i * KCHUNK;
    if (i + 1 < NCHUNK)                          // prefetch next chunk (async)
      tdm_issue(wt, col0, kb + KCHUNK, wv, (i & 1) ? lds0 : lds1);

    // ---- compute on current buffer while next DMA is in flight ----
    // A fragment (z, f16): lane m = lane&15, pieces [16h..] and [16h+8..]
    union { v16h v; v8h h[2]; } a;
    const _Float16* za = &lz[n * LZ_STRIDE + kb + (hf << 4)];
    a.h[0] = *(const v8h*)(za);
    a.h[1] = *(const v8h*)(za + 8);
    // B fragment from raw f32 chunk: element e -> K = e + 16h
    union { v16h v; _Float16 e[16]; } bf;
    const float* wr = &raw[i & 1][(hf * 16) * 128 + wv * 16 + n];
#pragma unroll
    for (int e = 0; e < 16; ++e) bf.e[e] = (_Float16)wr[e * 128];
    acc = __builtin_amdgcn_wmma_f32_16x16x32_f16(false, a.v, false, bf.v,
                                                 (short)0, acc, false, false);

    __syncthreads();                             // all waves done reading buf i
    if (i + 1 < NCHUNK) {
      __builtin_amdgcn_s_wait_tensorcnt(0);      // own slice of chunk i+1 done
      __syncthreads();                           // all 8 slices visible
    }
  }

  const int c = col0 + wv * 16 + n;
  const float bias = bt[c];
  if (hf == 0) {                                  // rows 0..7 = real batches
#pragma unroll
    for (int r = 0; r < 8; ++r)
      tab[(size_t)r * DTAB + c] = acc[r] + bias;
  }
}

// ============================================================================
// Kernel B: hash-grid gather (L2-resident tables) + 32->64->64->3 MLP in WMMA.
// grid 1024 blocks x 256 thr; 8 waves/block, 4 tiles of 16 pixels per wave.
// ============================================================================
__global__ __launch_bounds__(256)
void k_render(const float* __restrict__ tab,
              const float* __restrict__ w1, const float* __restrict__ b1,
              const float* __restrict__ w2, const float* __restrict__ b2,
              const float* __restrict__ w3, const float* __restrict__ b3,
              float* __restrict__ out) {
  __shared__ _Float16 w1T[64 * W1S];              // [n][k] transposed
  __shared__ _Float16 w2T[64 * W2S];
  __shared__ _Float16 w3T[16 * W3S];              // N padded 3->16
  __shared__ float lb1[64], lb2[64], lb3[16];
  __shared__ _Float16 lh[8 * 16 * HS];            // per-wave h bounce (16x64)

  const int tid = threadIdx.x;
  for (int idx = tid; idx < 32 * 64; idx += 256) {
    const int k = idx >> 6, nn = idx & 63;
    w1T[nn * W1S + k] = (_Float16)w1[idx];
  }
  for (int idx = tid; idx < 64 * 64; idx += 256) {
    const int k = idx >> 6, nn = idx & 63;
    w2T[nn * W2S + k] = (_Float16)w2[idx];
  }
  for (int idx = tid; idx < 16 * W3S; idx += 256) w3T[idx] = (_Float16)0.0f;
  if (tid < 64) { lb1[tid] = b1[tid]; lb2[tid] = b2[tid]; }
  if (tid < 16) lb3[tid] = (tid < 3) ? b3[tid] : 0.0f;
  __syncthreads();                                 // zero-fill before scatter
  for (int idx = tid; idx < 64 * 3; idx += 256) {
    const int k = idx / 3, nn = idx - 3 * k;
    w3T[nn * W3S + k] = (_Float16)w3[idx];
  }
  __syncthreads();

  const int lane = tid & 31, wv = tid >> 5;
  const int p = lane & 15, hf = lane >> 4;         // p: pixel / N-col, hf: half
  _Float16* hrow = &lh[(wv * 16) * HS];

  for (int it = 0; it < 4; ++it) {
    const int gt = blockIdx.x * 32 + wv * 4 + it;  // global 16-pixel tile id
    const int b  = gt >> 12;                       // 4096 tiles per batch
    const int pixbase = (gt & 4095) << 4;

    // ---- hash features: lane handles 8 of 16 levels for pixel (pixbase+p),
    // ---- filling exactly its A-fragment elements (K = 2*level + feat).
    const int pix = pixbase + p;
    const float x = ((float)(pix >> 8) + 0.5f) * (1.0f / 256.0f);
    const float y = ((float)(pix & 255) + 0.5f) * (1.0f / 256.0f);
    union { v16h v; _Float16 e[16]; } af;
#pragma unroll
    for (int pi = 0; pi < 8; ++pi) {
      const int lvl = 8 * hf + pi;
      const float res = exp2f(4.0f + (4.0f / 15.0f) * (float)lvl);
      float fx = x * res, fy = y * res;
      const float x0 = floorf(fx), y0 = floorf(fy);
      fx -= x0; fy -= y0;
      const unsigned xi = (unsigned)x0, yi = (unsigned)y0;
      const float* tb = tab + (((size_t)(b * NLVL + lvl)) << 13);
      const unsigned hy0 = yi * 2654435761u, hy1 = (yi + 1u) * 2654435761u;
      const unsigned i00 = (xi ^ hy0) & (TSIZE - 1);
      const unsigned i10 = ((xi + 1u) ^ hy0) & (TSIZE - 1);
      const unsigned i01 = (xi ^ hy1) & (TSIZE - 1);
      const unsigned i11 = ((xi + 1u) ^ hy1) & (TSIZE - 1);
      const float2 v00 = *(const float2*)(tb + 2u * i00);
      const float2 v10 = *(const float2*)(tb + 2u * i10);
      const float2 v01 = *(const float2*)(tb + 2u * i01);
      const float2 v11 = *(const float2*)(tb + 2u * i11);
      const float w00 = (1.0f - fx) * (1.0f - fy), w10 = fx * (1.0f - fy);
      const float w01 = (1.0f - fx) * fy,          w11 = fx * fy;
      af.e[2 * pi + 0] = (_Float16)(w00 * v00.x + w10 * v10.x + w01 * v01.x + w11 * v11.x);
      af.e[2 * pi + 1] = (_Float16)(w00 * v00.y + w10 * v10.y + w01 * v01.y + w11 * v11.y);
    }

    // ---- layer 1: (16x32) @ (32x64), relu, bounce to LDS ----
#pragma unroll
    for (int nt = 0; nt < 4; ++nt) {
      const float bb = lb1[nt * 16 + p];
      v8f c = { bb, bb, bb, bb, bb, bb, bb, bb };
      const v16h bw = ld16h(&w1T[(nt * 16 + p) * W1S + (hf << 4)]);
      c = __builtin_amdgcn_wmma_f32_16x16x32_f16(false, af.v, false, bw,
                                                 (short)0, c, false, false);
#pragma unroll
      for (int r = 0; r < 8; ++r) {
        const float vv = c[r] > 0.0f ? c[r] : 0.0f;
        hrow[(r + 8 * hf) * HS + nt * 16 + p] = (_Float16)vv;
      }
    }

    // ---- layer 2: (16x64) @ (64x64) ----
    union { v16h v; v8h h[2]; } a2[2];
#pragma unroll
    for (int j = 0; j < 2; ++j) {
      const _Float16* hp = &hrow[p * HS + 32 * j + 16 * hf];
      a2[j].h[0] = *(const v8h*)(hp);
      a2[j].h[1] = *(const v8h*)(hp + 8);
    }
#pragma unroll
    for (int nt = 0; nt < 4; ++nt) {
      const float bb = lb2[nt * 16 + p];
      v8f c = { bb, bb, bb, bb, bb, bb, bb, bb };
#pragma unroll
      for (int j = 0; j < 2; ++j) {
        const v16h bw = ld16h(&w2T[(nt * 16 + p) * W2S + 32 * j + (hf << 4)]);
        c = __builtin_amdgcn_wmma_f32_16x16x32_f16(false, a2[j].v, false, bw,
                                                   (short)0, c, false, false);
      }
#pragma unroll
      for (int r = 0; r < 8; ++r) {
        const float vv = c[r] > 0.0f ? c[r] : 0.0f;
        hrow[(r + 8 * hf) * HS + nt * 16 + p] = (_Float16)vv;
      }
    }

    // ---- layer 3: (16x64) @ (64x3 padded to 16), tanh, store ----
    union { v16h v; v8h h[2]; } a3[2];
#pragma unroll
    for (int j = 0; j < 2; ++j) {
      const _Float16* hp = &hrow[p * HS + 32 * j + 16 * hf];
      a3[j].h[0] = *(const v8h*)(hp);
      a3[j].h[1] = *(const v8h*)(hp + 8);
    }
    const float bb3 = lb3[p];
    v8f c3 = { bb3, bb3, bb3, bb3, bb3, bb3, bb3, bb3 };
#pragma unroll
    for (int j = 0; j < 2; ++j) {
      const v16h bw = ld16h(&w3T[p * W3S + 32 * j + (hf << 4)]);
      c3 = __builtin_amdgcn_wmma_f32_16x16x32_f16(false, a3[j].v, false, bw,
                                                  (short)0, c3, false, false);
    }
    if (p < 3) {                                   // channels 0..2 only
      float* ob = out + ((size_t)b * 3 + p) * IMGN + pixbase;
#pragma unroll
      for (int r = 0; r < 8; ++r) ob[r + 8 * hf] = tanhf(c3[r]);
    }
  }
}

// ============================================================================
extern "C" void kernel_launch(void* const* d_in, const int* in_sizes, int n_in,
                              void* d_out, int out_size, void* d_ws, size_t ws_size,
                              hipStream_t stream) {
  (void)in_sizes; (void)n_in; (void)out_size; (void)ws_size;
  const float* z  = (const float*)d_in[0];
  const float* wt = (const float*)d_in[1];
  const float* bt = (const float*)d_in[2];
  const float* w1 = (const float*)d_in[3];
  const float* b1 = (const float*)d_in[4];
  const float* w2 = (const float*)d_in[5];
  const float* b2 = (const float*)d_in[6];
  const float* w3 = (const float*)d_in[7];
  const float* b3 = (const float*)d_in[8];
  float* tab = (float*)d_ws;                      // 8 x 131072 f32 = 4 MB

  k_tables<<<1024, 256, 0, stream>>>(z, wt, bt, tab);
  k_render<<<1024, 256, 0, stream>>>(tab, w1, b1, w2, b2, w3, b3, (float*)d_out);
}
